// DecoderLayer_73478300500220
// MI455X (gfx1250) — matrix operand order repaired and checked by reading, not compiled, not run on hardware
//
#include <hip/hip_runtime.h>
#include <hip/hip_bf16.h>

// ---------------------------------------------------------------------------
// Decoder layer for MI455X (gfx1250, wave32, WMMA, TDM).
// B=2048, S=128, D_MODEL=50 (pad 64), H=5, DEPTH=10 (pad 16), DFF=64.
// v_wmma_f32_16x16x32_f16 everywhere; fragments = contiguous 16B loads
// (weights pre-swizzled); fused QKV projection; attention K-tile via TDM.
// ---------------------------------------------------------------------------

typedef __attribute__((ext_vector_type(16))) _Float16 v16h;
typedef __attribute__((ext_vector_type(8)))  _Float16 v8h;
typedef __attribute__((ext_vector_type(8)))  float    v8f;
typedef unsigned int u32x4 __attribute__((ext_vector_type(4)));
typedef int          i32x8 __attribute__((ext_vector_type(8)));
typedef int          i32x4 __attribute__((ext_vector_type(4)));

#define BS_TOTAL (2048 * 128)
#define S_LEN    128
#define NBATCH   2048
#define NH       5

enum { EP_SILU = 1, EP_LN_R32_O16 = 2, EP_LN_R16_O16 = 3, EP_LN_R16_O32 = 4 };

__device__ inline v8f wmma16(v16h a, v16h b, v8f c) {
  return __builtin_amdgcn_wmma_f32_16x16x32_f16(false, a, false, b, (short)0, c, false, false);
}

__device__ inline v16h cat8(v8h lo, v8h hi) {
  return __builtin_shufflevector(lo, hi, 0, 1, 2, 3, 4, 5, 6, 7, 8, 9, 10, 11, 12, 13, 14, 15);
}

__device__ inline v8h zero8h() {
  v8h z = {(_Float16)0.0f, (_Float16)0.0f, (_Float16)0.0f, (_Float16)0.0f,
           (_Float16)0.0f, (_Float16)0.0f, (_Float16)0.0f, (_Float16)0.0f};
  return z;
}

// A fragment from row-major (ld=64) f16: k=k0+h*16+j
__device__ inline v16h load_a_g(const _Float16* X, int row, int k0, int hh) {
  const _Float16* p = X + (size_t)row * 64 + k0 + hh * 16;
  return cat8(*(const v8h*)p, *(const v8h*)(p + 8));
}

// B fragment from swizzled weights: tile t=(ktile*4+nt), lane n -> 16 contiguous f16
__device__ inline v16h load_b_swz(const _Float16* W, int ktile, int nt, int n) {
  const _Float16* p = W + ((((ktile << 2) + nt) * 16 + n) << 4);
  return cat8(*(const v8h*)p, *(const v8h*)(p + 8));
}

// Softmax-prob A fragment: fp32 LDS (ld=128) -> f16 fragment
__device__ inline v16h load_a_p(const float* sS, int row, int k0, int hh) {
  const float* p = sS + row * 128 + k0 + hh * 16;
  v8f lo = *(const v8f*)p;
  v8f hi = *(const v8f*)(p + 8);
  return cat8(__builtin_convertvector(lo, v8h), __builtin_convertvector(hi, v8h));
}

// 1-D TDM copy: nelem16 f16 elements global -> LDS (contiguous)
__device__ inline void tdm_load_1d(unsigned int lds_off, const void* gptr, unsigned int nelem16) {
  unsigned long long ga = (unsigned long long)(uintptr_t)gptr;
  u32x4 g0;
  g0[0] = 1u;                                   // count=1 valid descriptor
  g0[1] = lds_off;                              // lds_addr
  g0[2] = (unsigned int)ga;                     // global_addr[31:0]
  g0[3] = (unsigned int)((ga >> 32) & 0x1FFFFFFu) | (2u << 30);  // addr[56:32] | type=2
  i32x8 g1;
  g1[0] = 0x00010000;                           // data_size=1 (2 bytes)
  g1[1] = (int)(nelem16 << 16);                 // tensor_dim0[15:0]
  g1[2] = 0x00010000;                           // tensor_dim1 = 1
  g1[3] = (int)(nelem16 << 16);                 // tile_dim0
  g1[4] = 0;
  g1[5] = (int)nelem16;                         // tensor_dim0_stride
  g1[6] = 0;
  g1[7] = 0;
  i32x4 gz4; gz4[0] = 0; gz4[1] = 0; gz4[2] = 0; gz4[3] = 0;
  i32x8 gz8; gz8[0] = 0; gz8[1] = 0; gz8[2] = 0; gz8[3] = 0;
  gz8[4] = 0; gz8[5] = 0; gz8[6] = 0; gz8[7] = 0;
  __builtin_amdgcn_tensor_load_to_lds(g0, g1, gz4, gz4, gz8, 0);
}

// ---------------------------------------------------------------------------
// Prep kernels
// ---------------------------------------------------------------------------
__global__ void pad_x_kernel(const float* __restrict__ src, _Float16* __restrict__ dst) {
  int gid = blockIdx.x * blockDim.x + threadIdx.x;   // BS_TOTAL*64 threads
  int r = gid >> 6, c = gid & 63;
  dst[gid] = (c < 50) ? (_Float16)src[r * 50 + c] : (_Float16)0.0f;
}

// Weights stored fragment-swizzled: index = ((ktile*4+nt)*16 + n)*16 + j,
// value = W[ktile*16+j][nt*16+n] (zero-padded to 64x64)
__global__ void prep_w_kernel(const float* wq, const float* bq, const float* wk, const float* bk,
                              const float* wv, const float* bv, const float* wo, const float* bo,
                              const float* w1, const float* b1, const float* w2, const float* b2,
                              _Float16* __restrict__ W16, float* __restrict__ B64) {
  int tid = threadIdx.x;
  const float* wsrc[6] = {wq, wk, wv, wo, w1, w2};
  const int kd[6] = {50, 50, 50, 50, 50, 64};
  const int nd[6] = {50, 50, 50, 50, 64, 50};
  for (int w = 0; w < 6; ++w) {
    const float* s = wsrc[w];
    _Float16* d = W16 + w * 4096;
    int kk = kd[w], nn = nd[w];
    for (int i = tid; i < 4096; i += 256) {
      int t = i >> 8, n = (i >> 4) & 15, j = i & 15;
      int k = (t >> 2) * 16 + j, c = (t & 3) * 16 + n;
      d[i] = (k < kk && c < nn) ? (_Float16)s[k * nn + c] : (_Float16)0.0f;
    }
  }
  const float* bsrc[6] = {bq, bk, bv, bo, b1, b2};
  const int bl[6] = {50, 50, 50, 50, 64, 50};
  for (int w = 0; w < 6; ++w)
    for (int i = tid; i < 64; i += 256)
      B64[w * 64 + i] = (i < bl[w]) ? bsrc[w][i] : 0.0f;
}

// ---------------------------------------------------------------------------
// Fused QKV projection: one block = one batch (128 rows), 8 waves.
// q,k from Xqk; v from Xv (v_same=1 if Xv==Xqk). 24 WMMAs/wave.
// Epilogue scatters into (B,H,S,16) head layout and zeros the d>=10 pads.
// ---------------------------------------------------------------------------
__global__ __launch_bounds__(256)
void gemm_qkv3(const _Float16* __restrict__ Xqk, const _Float16* __restrict__ Xv, int v_same,
               const _Float16* __restrict__ Wqkv, const float* __restrict__ Bqkv,
               _Float16* __restrict__ Qh, _Float16* __restrict__ Kh, _Float16* __restrict__ Vh) {
  const int tid = threadIdx.x;
  const int bidx = blockIdx.x;               // batch index; rows [bidx*128, +128)
  const int row_base = bidx * 128;
  const int wave = tid >> 5;
  const int lane = tid & 31;
  const int row0 = wave * 16;
  const int hh = lane >> 4;
  const int n = lane & 15;

  v8f accq[4], acck[4], accv[4];
#pragma unroll
  for (int t = 0; t < 4; ++t)
#pragma unroll
    for (int e = 0; e < 8; ++e) { accq[t][e] = 0.0f; acck[t][e] = 0.0f; accv[t][e] = 0.0f; }

#pragma unroll
  for (int kc = 0; kc < 2; ++kc) {
    v16h aqk = load_a_g(Xqk, row_base + row0 + n, kc * 32, hh);
    v16h av = v_same ? aqk : load_a_g(Xv, row_base + row0 + n, kc * 32, hh);
#pragma unroll
    for (int nt = 0; nt < 4; ++nt) {
      accq[nt] = wmma16(aqk, load_b_swz(Wqkv, kc * 2 + hh, nt, n), accq[nt]);
      acck[nt] = wmma16(aqk, load_b_swz(Wqkv + 4096, kc * 2 + hh, nt, n), acck[nt]);
      accv[nt] = wmma16(av, load_b_swz(Wqkv + 8192, kc * 2 + hh, nt, n), accv[nt]);
    }
  }

#pragma unroll
  for (int nt = 0; nt < 4; ++nt) {
    int c = nt * 16 + n;
    if (c < 50) {
      int head = c / 10, d = c % 10;
      float bcq = Bqkv[c], bck = Bqkv[64 + c], bcv = Bqkv[128 + c];
#pragma unroll
      for (int r = 0; r < 8; ++r) {
        int s = row0 + r + 8 * hh;
        size_t o = ((((size_t)bidx * NH + head) * S_LEN + s) << 4) + d;
        Qh[o] = (_Float16)(accq[nt][r] + bcq);
        Kh[o] = (_Float16)(acck[nt][r] + bck);
        Vh[o] = (_Float16)(accv[nt][r] + bcv);
      }
    }
  }
  // zero depth pads d=10..15 (dwords 5..7 of each 32B head row) for this batch
  for (int i = tid; i < NH * S_LEN * 3; i += 256) {
    int hs = i / 3, w = i % 3;
    size_t rowoff = (((size_t)bidx * NH + (hs >> 7)) * S_LEN + (hs & 127)) << 4;
    ((unsigned int*)(Qh + rowoff))[5 + w] = 0u;
    ((unsigned int*)(Kh + rowoff))[5 + w] = 0u;
    ((unsigned int*)(Vh + rowoff))[5 + w] = 0u;
  }
}

// ---------------------------------------------------------------------------
// LDS-free 128x64 GEMM (K=64) with fused epilogues (silu / residual+LN).
// 256 threads = 8 waves; wave w owns rows [w*16,w*16+16) x all 64 cols.
// ---------------------------------------------------------------------------
__global__ __launch_bounds__(256)
void gemm64(const _Float16* __restrict__ X, const _Float16* __restrict__ W,
            const float* __restrict__ bias, int mode,
            _Float16* __restrict__ out16, float* __restrict__ out32,
            const float* __restrict__ res32, const _Float16* __restrict__ res16,
            const float* __restrict__ lng, const float* __restrict__ lnb) {
  const int tid = threadIdx.x;
  const int row_base = blockIdx.x * 128;
  const int wave = tid >> 5;
  const int lane = tid & 31;
  const int row0 = wave * 16;
  const int hh = lane >> 4;
  const int n = lane & 15;

  v8f acc[4];
#pragma unroll
  for (int t = 0; t < 4; ++t)
#pragma unroll
    for (int e = 0; e < 8; ++e) acc[t][e] = 0.0f;

#pragma unroll
  for (int kc = 0; kc < 2; ++kc) {
    v16h a = load_a_g(X, row_base + row0 + n, kc * 32, hh);
#pragma unroll
    for (int nt = 0; nt < 4; ++nt) {
      v16h b = load_b_swz(W, kc * 2 + hh, nt, n);
      acc[nt] = wmma16(a, b, acc[nt]);
    }
  }

  if (mode == EP_SILU) {
#pragma unroll
    for (int nt = 0; nt < 4; ++nt) {
      int c = nt * 16 + n;
      float bc = bias[c];
#pragma unroll
      for (int r = 0; r < 8; ++r) {
        int gr = row_base + row0 + r + 8 * hh;
        float v = acc[nt][r] + bc;
        out16[(size_t)gr * 64 + c] = (_Float16)(v / (1.0f + __expf(-v)));
      }
    }
  } else {
    float vals[4][8];
#pragma unroll
    for (int nt = 0; nt < 4; ++nt) {
      int c = nt * 16 + n;
      float bc = bias[c];
#pragma unroll
      for (int r = 0; r < 8; ++r) {
        int gr = row_base + row0 + r + 8 * hh;
        float v = acc[nt][r] + bc;
        if (mode == EP_LN_R32_O16) {
          if (c < 50) v += res32[(size_t)gr * 50 + c];
        } else {
          v += (float)res16[(size_t)gr * 64 + c];
        }
        vals[nt][r] = v;   // exactly zero for c>=50 pads -> harmless in sums
      }
    }
#pragma unroll
    for (int r = 0; r < 8; ++r) {
      float s1 = vals[0][r] + vals[1][r] + vals[2][r] + vals[3][r];
      float s2 = vals[0][r] * vals[0][r] + vals[1][r] * vals[1][r] +
                 vals[2][r] * vals[2][r] + vals[3][r] * vals[3][r];
#pragma unroll
      for (int m = 1; m <= 8; m <<= 1) {   // reduce across the 16 lanes of each half
        s1 += __shfl_xor(s1, m, 32);
        s2 += __shfl_xor(s2, m, 32);
      }
      float mu = s1 * (1.0f / 50.0f);
      float var = s2 * (1.0f / 50.0f) - mu * mu;
      float rstd = rsqrtf(var + 1e-6f);
      int gr = row_base + row0 + r + 8 * hh;
#pragma unroll
      for (int nt = 0; nt < 4; ++nt) {
        int c = nt * 16 + n;
        if (mode == EP_LN_R16_O32) {
          if (c < 50) out32[(size_t)gr * 50 + c] = (vals[nt][r] - mu) * rstd * lng[c] + lnb[c];
        } else {
          float y = (c < 50) ? (vals[nt][r] - mu) * rstd * lng[c] + lnb[c] : 0.0f;
          out16[(size_t)gr * 64 + c] = (_Float16)y;
        }
      }
    }
  }
}

// ---------------------------------------------------------------------------
// Attention: block = (batch, head, 64-query chunk); 128 threads = 4 waves.
// K tile via TDM; V swizzled into fragment order; softmax in LDS fp32.
// head==0 blocks also zero attn_cat's column pads (dwords 25..31 per row).
// ---------------------------------------------------------------------------
__global__ __launch_bounds__(128)
void attn64(const _Float16* __restrict__ Q, const _Float16* __restrict__ K,
            const _Float16* __restrict__ V, const float* __restrict__ mask,
            _Float16* __restrict__ attn_cat) {
  __shared__ _Float16 sK[128 * 16];    // raw (s,d) rows, d pad already zero
  __shared__ _Float16 sVs[128 * 16];   // swizzled: ((kt*16+n)*16 + j) = V[kt*16+j][n]
  __shared__ float sS[64 * 128];

  const int tid = threadIdx.x;
  const int idx = blockIdx.x;
  const int qc = idx & 1;
  const int bh = idx >> 1;
  const int head = bh % NH;
  const int b = bh / NH;
  const size_t base = ((size_t)bh * S_LEN) << 4;

  if (tid == 0)
    tdm_load_1d((unsigned int)(uintptr_t)&sK[0], K + base, 128 * 16);
  for (int i = tid; i < 128 * 16; i += 128) {
    int j = i & 15, nn = (i >> 4) & 15, kt = i >> 8;
    sVs[i] = V[base + (size_t)(kt * 16 + j) * 16 + nn];
  }
  if (head == 0) {  // zero attn_cat cols 50..63 for this (b, qc) slab
    for (int i = tid; i < 64 * 7; i += 128) {
      size_t grow = (size_t)b * S_LEN + qc * 64 + i / 7;
      ((unsigned int*)(attn_cat + grow * 64))[25 + i % 7] = 0u;
    }
  }
  if (tid == 0) __builtin_amdgcn_s_wait_tensorcnt(0);
  __syncthreads();

  const int wave = tid >> 5;
  const int lane = tid & 31;
  const int row0 = wave * 16;
  const int hh = lane >> 4;
  const int n = lane & 15;

  // logits = Q K^T * scale + mask * -1e9 ; Q A-fragment straight from global
  const _Float16* qp = Q + base + (size_t)(qc * 64 + row0 + n) * 16;
  v16h aq = cat8(*(const v8h*)qp, *(const v8h*)(qp + 8));      // logical K-dim = 16
  const _Float16 bsel = (hh == 0) ? (_Float16)1.0f : (_Float16)0.0f;
#pragma unroll
  for (int ct = 0; ct < 8; ++ct) {
    const _Float16* kp = sK + (ct * 16 + n) * 16;
    v16h bk = cat8(*(const v8h*)kp, *(const v8h*)(kp + 8));
    bk *= bsel;                                   // zero the K>=16 half
    v8f sc;
#pragma unroll
    for (int e = 0; e < 8; ++e) sc[e] = 0.0f;
    sc = wmma16(aq, bk, sc);
#pragma unroll
    for (int r = 0; r < 8; ++r) {
      int qr = row0 + r + 8 * hh;
      int kc = ct * 16 + n;
      sS[qr * 128 + kc] = sc[r] * 0.316227766f + mask[(qc * 64 + qr) * 128 + kc] * (-1e9f);
    }
  }
  __syncthreads();

  if (tid < 64) {   // row softmax
    float* row = sS + tid * 128;
    float mx = -3.4e38f;
    for (int c = 0; c < 128; ++c) mx = fmaxf(mx, row[c]);
    float sum = 0.0f;
    for (int c = 0; c < 128; ++c) { float e = __expf(row[c] - mx); row[c] = e; sum += e; }
    float inv = 1.0f / sum;
    for (int c = 0; c < 128; ++c) row[c] *= inv;
  }
  __syncthreads();

  // out = P @ V (K=128 -> 4 WMMA chunks)
  v8f o;
#pragma unroll
  for (int e = 0; e < 8; ++e) o[e] = 0.0f;
#pragma unroll
  for (int kc = 0; kc < 4; ++kc) {
    v16h ap = load_a_p(sS, row0 + n, kc * 32, hh);
    const _Float16* vp = sVs + (((kc * 2 + hh) * 16 + n) << 4);
    v16h bv = cat8(*(const v8h*)vp, *(const v8h*)(vp + 8));
    o = wmma16(ap, bv, o);
  }
  if (n < 10) {
#pragma unroll
    for (int r = 0; r < 8; ++r) {
      int qr = row0 + r + 8 * hh;
      size_t grow = (size_t)b * S_LEN + qc * 64 + qr;
      attn_cat[grow * 64 + head * 10 + n] = (_Float16)o[r];
    }
  }
}

// ---------------------------------------------------------------------------
extern "C" void kernel_launch(void* const* d_in, const int* in_sizes, int n_in,
                              void* d_out, int out_size, void* d_ws, size_t ws_size,
                              hipStream_t stream) {
  const float* dec = (const float*)d_in[0];
  const float* enc = (const float*)d_in[1];
  const float* mask_self = (const float*)d_in[2];
  const float* mask_cross = (const float*)d_in[3];
  const float* wq = (const float*)d_in[4];  const float* bq = (const float*)d_in[5];
  const float* wk = (const float*)d_in[6];  const float* bk = (const float*)d_in[7];
  const float* wv = (const float*)d_in[8];  const float* bv = (const float*)d_in[9];
  const float* wo = (const float*)d_in[10]; const float* bo = (const float*)d_in[11];
  const float* w1 = (const float*)d_in[12]; const float* b1 = (const float*)d_in[13];
  const float* w2 = (const float*)d_in[14]; const float* b2 = (const float*)d_in[15];
  const float* ln1g = (const float*)d_in[16]; const float* ln1b = (const float*)d_in[17];
  const float* ln2g = (const float*)d_in[18]; const float* ln2b = (const float*)d_in[19];
  const float* ln3g = (const float*)d_in[20]; const float* ln3b = (const float*)d_in[21];
  float* out = (float*)d_out;

  char* ws = (char*)d_ws;
  _Float16* W16 = (_Float16*)ws;            // 6 x 64x64 f16 (fragment-swizzled)
  float* B64 = (float*)(ws + 6 * 8192);     // 6 x 64 fp32 padded biases

  const size_t SZ16 = (size_t)BS_TOTAL * 64 * sizeof(_Float16);              // 32 MB
  const size_t SZH  = (size_t)NBATCH * NH * S_LEN * 16 * sizeof(_Float16);   // 40 MB
  size_t off = 65536;
  _Float16* Xdec = (_Float16*)(ws + off); off += SZ16;   // reused as FFN hidden
  _Float16* Xenc = (_Float16*)(ws + off); off += SZ16;   // reused as ca16
  _Float16* Qh   = (_Float16*)(ws + off); off += SZH;
  _Float16* Kh   = (_Float16*)(ws + off); off += SZH;
  _Float16* Vh   = (_Float16*)(ws + off); off += SZH;
  _Float16* acat = (_Float16*)(ws + off); off += SZ16;
  _Float16* sa16 = (_Float16*)(ws + off); off += SZ16;
  _Float16* h16  = Xdec;
  _Float16* ca16 = Xenc;

  const _Float16 *Wqkv = W16, *Wo16 = W16 + 12288, *W116 = W16 + 16384, *W216 = W16 + 20480;
  const float *Bqkv = B64, *Bo = B64 + 192, *B1 = B64 + 256, *B2 = B64 + 320;

  prep_w_kernel<<<1, 256, 0, stream>>>(wq, bq, wk, bk, wv, bv, wo, bo, w1, b1, w2, b2, W16, B64);
  pad_x_kernel<<<BS_TOTAL * 64 / 256, 256, 0, stream>>>(dec, Xdec);
  pad_x_kernel<<<BS_TOTAL * 64 / 256, 256, 0, stream>>>(enc, Xenc);

  const int GB = BS_TOTAL / 128;   // 2048 blocks (one batch each)
  // ---- self attention: q=k=v=dec ----
  gemm_qkv3<<<GB, 256, 0, stream>>>(Xdec, Xdec, 1, Wqkv, Bqkv, Qh, Kh, Vh);
  attn64<<<NBATCH * NH * 2, 128, 0, stream>>>(Qh, Kh, Vh, mask_self, acat);
  gemm64<<<GB, 256, 0, stream>>>(acat, Wo16, Bo, EP_LN_R32_O16, sa16, nullptr, dec, nullptr, ln1g, ln1b);
  // ---- cross attention: v=sa, k=q=enc (shared weights) ----
  gemm_qkv3<<<GB, 256, 0, stream>>>(Xenc, sa16, 0, Wqkv, Bqkv, Qh, Kh, Vh);
  attn64<<<NBATCH * NH * 2, 128, 0, stream>>>(Qh, Kh, Vh, mask_cross, acat);
  gemm64<<<GB, 256, 0, stream>>>(acat, Wo16, Bo, EP_LN_R16_O16, ca16, nullptr, nullptr, sa16, ln2g, ln2b);
  // ---- FFN ----
  gemm64<<<GB, 256, 0, stream>>>(ca16, W116, B1, EP_SILU, h16, nullptr, nullptr, nullptr, nullptr, nullptr);
  gemm64<<<GB, 256, 0, stream>>>(h16, W216, B2, EP_LN_R16_O32, nullptr, out, nullptr, ca16, ln3g, ln3b);
}
